// TemporalSAE_47579647705686
// MI455X (gfx1250) — matrix operand order repaired and checked by reading, not compiled, not run on hardware
//
#include <hip/hip_runtime.h>
#include <hip/hip_bf16.h>
#include <math.h>

typedef unsigned short u16;
typedef __attribute__((ext_vector_type(16))) __bf16 v16bf;
typedef __attribute__((ext_vector_type(8)))  float  v8f;

#define B_    4
#define L_    1024
#define DIN_  768
#define W_    12288
#define H_    8
#define NL_   2
#define DB_   192
#define HD_   24
#define TOPK_ 32
#define M_    (B_*L_)     // 4096 rows
#define BH_   (B_*H_)     // 32 attention batches
#define HP_   32          // head dim padded for QK^T (K dim)
#define VP_   64          // head dim padded for PV (N dim)
#define LAM_  (1.0f/(4.0f*(float)DIN_))
#define EPS_  1e-6f

__device__ __forceinline__ u16 f2bf(float f) {
  union { float f; unsigned u; } v; v.f = f;
  unsigned r = v.u + 0x7FFFu + ((v.u >> 16) & 1u);   // round-to-nearest-even
  return (u16)(r >> 16);
}

// ---------------------------------------------------------------------------
// Generic batched bf16 WMMA GEMM: C = act(alpha * A@B + bias)
// A: [batch?][M,K] bf16 row-major, B: [batch?][K,N] bf16 row-major
// Block: 256 threads (8 waves). Block tile 128 x (64*NB), wave tile 32x(32*NB),
// K stepped by 32. Double-buffered LDS; global loads for tile t+1 issued
// before the WMMAs on tile t (latency hidden behind matrix ops).
// Requires M%128==0, N%(64*NB)==0, K%32==0.
// ---------------------------------------------------------------------------
template<int NB>
__global__ __launch_bounds__(256) void gemm_bf16_wmma_kernel(
    const u16* __restrict__ A, const u16* __restrict__ Bm,
    const float* __restrict__ bias, float* __restrict__ Cf, u16* __restrict__ Cbf,
    int M, int N, int K,
    long long sA, long long sB, long long sC,
    float alpha, int relu)
{
  // 40 u16 = 80B row stride: 16B-aligned b128 fragment reads, 20-bank spread
  __shared__ __align__(16) u16 As[2][128][40];       // [m][k]
  __shared__ __align__(16) u16 Bs[2][64 * NB][40];   // [n][k] (transposed)

  const int tid  = threadIdx.x;
  const int lane = tid & 31, wave = tid >> 5;
  const int wr = wave >> 1, wc = wave & 1;           // wave tile origin
  const long long bz = blockIdx.z;
  const u16* Ab = A  + bz * sA;
  const u16* Bb = Bm + bz * sB;
  const int m0 = blockIdx.y * 128, n0 = blockIdx.x * (64 * NB);

  v8f c[2][2 * NB];
#pragma unroll
  for (int i = 0; i < 2; i++)
#pragma unroll
    for (int j = 0; j < 2 * NB; j++)
#pragma unroll
      for (int e = 0; e < 8; e++) c[i][j][e] = 0.0f;

  // staging assignments
  const int arow = tid >> 1, ahalf = tid & 1;        // A: 128 rows x 32 k
  const int NGRP = 8 * NB;                           // B n-groups of 8
  const int btask = tid;                             // B: 16 kpairs x NGRP
  const int bkp = btask / NGRP, bng = (btask % NGRP) * 8;
  const bool bact = (btask < 16 * NGRP);

  uint4 ar0, ar1, br0, br1;

  auto gload = [&](int k0) {
    const u16* asrc = Ab + (long long)(m0 + arow) * K + k0 + ahalf * 16;
    ar0 = ((const uint4*)asrc)[0];
    ar1 = ((const uint4*)asrc)[1];
    if (bact) {
      const u16* bsrc = Bb + (long long)(k0 + 2 * bkp) * N + n0 + bng;
      br0 = *(const uint4*)bsrc;
      br1 = *(const uint4*)(bsrc + N);
    }
  };
  auto sstore = [&](int buf) {
    *(uint4*)&As[buf][arow][ahalf * 16]     = ar0;
    *(uint4*)&As[buf][arow][ahalf * 16 + 8] = ar1;
    if (bact) {
      const u16* e0 = (const u16*)&br0;
      const u16* e1 = (const u16*)&br1;
#pragma unroll
      for (int i2 = 0; i2 < 8; i2++) {   // pack 2 k-rows -> one b32 store
        unsigned pk = (unsigned)e0[i2] | ((unsigned)e1[i2] << 16);
        *(unsigned*)&Bs[buf][bng + i2][2 * bkp] = pk;
      }
    }
  };

  gload(0);
  int buf = 0;
  const int ml = lane & 15, abase = (lane >> 4) * 16; // A frag: lane layout
  const int nl = lane & 15, kb   = (lane >> 4) * 16;  // B frag: lane layout

  for (int k0 = 0; k0 < K; k0 += 32) {
    sstore(buf);
    __syncthreads();
    if (k0 + 32 < K) {
      gload(k0 + 32);  // in flight while WMMAs below execute
      __builtin_prefetch(Ab + (long long)(m0 + arow) * K + k0 + 64 + ahalf * 16, 0, 1);
      __builtin_prefetch(Bb + (long long)(k0 + 64 + 2 * bkp) * N + n0 + bng, 0, 1);
    }

    union Frag { v16bf v; uint4 u[2]; } a[2], b[2 * NB];
#pragma unroll
    for (int i = 0; i < 2; i++) {
      a[i].u[0] = *(const uint4*)&As[buf][wr * 32 + i * 16 + ml][abase];
      a[i].u[1] = *(const uint4*)&As[buf][wr * 32 + i * 16 + ml][abase + 8];
    }
#pragma unroll
    for (int j = 0; j < 2 * NB; j++) {
      b[j].u[0] = *(const uint4*)&Bs[buf][wc * (32 * NB) + j * 16 + nl][kb];
      b[j].u[1] = *(const uint4*)&Bs[buf][wc * (32 * NB) + j * 16 + nl][kb + 8];
    }
#pragma unroll
    for (int i = 0; i < 2; i++)
#pragma unroll
      for (int j = 0; j < 2 * NB; j++)
        c[i][j] = __builtin_amdgcn_wmma_f32_16x16x32_bf16(
            false, a[i].v, false, b[j].v, (short)0, c[i][j], false, false);
    buf ^= 1;
  }

  // Epilogue: C layout — lanes 0-15: M=e, lanes 16-31: M=8+e; N = lane&15.
  const int crow = (lane >> 4) * 8, cn = lane & 15;
#pragma unroll
  for (int i = 0; i < 2; i++) {
#pragma unroll
    for (int j = 0; j < 2 * NB; j++) {
      int n = n0 + wc * (32 * NB) + j * 16 + cn;
      float bvv = bias ? bias[n] : 0.0f;
#pragma unroll
      for (int e = 0; e < 8; e++) {
        int m = m0 + wr * 32 + i * 16 + crow + e;
        float v = c[i][j][e] * alpha + bvv;
        if (relu) v = v > 0.0f ? v : 0.0f;
        long long off = bz * sC + (long long)m * N + n;
        if (Cf)  Cf[off]  = v;
        if (Cbf) Cbf[off] = f2bf(v);
      }
    }
  }
}

// ---------------------------------------------------------------------------
// Elementwise / helper kernels
// ---------------------------------------------------------------------------
__global__ void fill_zero_kernel(float* p, long long n) {
  long long i  = (long long)blockIdx.x * blockDim.x + threadIdx.x;
  long long st = (long long)gridDim.x * blockDim.x;
  for (; i < n; i += st) p[i] = 0.0f;
}

__global__ void prep_x_kernel(const float* x, const float* b, float* xs, u16* xsbf,
                              long long n, int din) {
  long long i  = (long long)blockIdx.x * blockDim.x + threadIdx.x;
  long long st = (long long)gridDim.x * blockDim.x;
  for (; i < n; i += st) {
    float v = x[i] - b[i % din];
    xs[i] = v; xsbf[i] = f2bf(v);
  }
}

__global__ void cvt_bf16_kernel(const float* in, u16* out, long long n) {
  long long i  = (long long)blockIdx.x * blockDim.x + threadIdx.x;
  long long st = (long long)gridDim.x * blockDim.x;
  for (; i < n; i += st) out[i] = f2bf(in[i]);
}

// out[C x R] = transpose of in[R x C]; output-contiguous indexing
__global__ void transpose_cvt_kernel(const float* in, u16* out, int R, int C) {
  long long n  = (long long)R * C;
  long long i  = (long long)blockIdx.x * blockDim.x + threadIdx.x;
  long long st = (long long)gridDim.x * blockDim.x;
  for (; i < n; i += st) {
    long long cc = i / R, rr = i % R;
    out[i] = f2bf(in[rr * (long long)C + cc]);
  }
}

// z_ctx[b,l,:] = (l==0) ? 0 : z_in[b,l-1,:]   (bf16, uint4 copies)
__global__ void shift_rows_kernel(const u16* zin, u16* zctx) {
  int r = blockIdx.y;
  int l = r & (L_ - 1);
  uint4* dst = (uint4*)(zctx + (long long)r * W_);
  const uint4* src = (const uint4*)(zin + (long long)(r - 1) * W_);
  int nvec = W_ / 8;
  for (int i = blockIdx.x * blockDim.x + threadIdx.x; i < nvec;
       i += gridDim.x * blockDim.x) {
    uint4 v;
    if (l == 0) { v.x = v.y = v.z = v.w = 0u; } else v = src[i];
    dst[i] = v;
  }
}

// Split heads + pad: Qp[bh,l,32], Kt[bh,32,l] (transposed), Vp[bh,l,64]
__global__ void pack_qkv_kernel(const float* qf, const float* kf, const float* vf,
                                u16* Qp, u16* Kt, u16* Vp) {
  long long i = (long long)blockIdx.x * blockDim.x + threadIdx.x;
  if (i >= (long long)BH_ * L_ * VP_) return;
  int d  = (int)(i & 63);
  int l  = (int)((i >> 6) & (L_ - 1));
  int bh = (int)(i >> 16);              // L_*VP_ == 65536
  int b = bh >> 3, h = bh & 7;
  long long rowoff = (long long)(b * L_ + l) * DB_ + h * HD_;
  float vv = (d < HD_) ? vf[rowoff + d] : 0.0f;
  Vp[i] = f2bf(vv);
  if (d < HP_) {
    float qv = (d < HD_) ? qf[rowoff + d] : 0.0f;
    float kv = (d < HD_) ? kf[rowoff + d] : 0.0f;
    Qp[((long long)bh * L_ + l) * HP_ + d] = f2bf(qv);
    Kt[((long long)bh * HP_ + d) * L_ + l] = f2bf(kv);
  }
}

__global__ __launch_bounds__(256) void softmax_causal_kernel(
    const float* scores, u16* probs) {
  int q = blockIdx.x, bh = blockIdx.y, tid = threadIdx.x;
  const float* row = scores + ((long long)bh * L_ + q) * L_;
  u16* orow = probs + ((long long)bh * L_ + q) * L_;
  __shared__ float red[256];
  const float scale = 0.20412414523193154f;  // 1/sqrt(24)
  float mx = -1e30f;
  for (int cIdx = tid; cIdx <= q; cIdx += 256) {
    float v = row[cIdx] * scale; mx = v > mx ? v : mx;
  }
  red[tid] = mx; __syncthreads();
  for (int s = 128; s > 0; s >>= 1) {
    if (tid < s && red[tid + s] > red[tid]) red[tid] = red[tid + s];
    __syncthreads();
  }
  mx = red[0]; __syncthreads();
  float sum = 0.0f;
  for (int cIdx = tid; cIdx <= q; cIdx += 256)
    sum += __expf(row[cIdx] * scale - mx);
  red[tid] = sum; __syncthreads();
  for (int s = 128; s > 0; s >>= 1) {
    if (tid < s) red[tid] += red[tid + s];
    __syncthreads();
  }
  float inv = 1.0f / red[0];
  for (int cIdx = tid; cIdx < L_; cIdx += 256) {
    float p = (cIdx <= q) ? __expf(row[cIdx] * scale - mx) * inv : 0.0f;
    orow[cIdx] = f2bf(p);
  }
}

// Opad[bh,l,64] -> o_bf[4096,192] (drop padding)
__global__ void unpack_o_kernel(const float* Opad, u16* obf) {
  long long i = (long long)blockIdx.x * blockDim.x + threadIdx.x;
  if (i >= (long long)M_ * DB_) return;
  int col = (int)(i % DB_);
  int row = (int)(i / DB_);
  int h = col / HD_, d = col % HD_;
  int b = row >> 10, l = row & (L_ - 1);
  obf[i] = f2bf(Opad[((long long)(b * H_ + h) * L_ + l) * VP_ + d]);
}

// Per-row: s1=dot(Dz,x), s2=dot(Dz,Dz); scale=s1/(||Dz||+eps)^2; x -= scale*Dz
__global__ __launch_bounds__(256) void proj_update_kernel(
    const float* Dz, float* xs, u16* xsbf, float* scale) {
  int r = blockIdx.x, tid = threadIdx.x;
  __shared__ float r1[256], r2[256];
  const float* dz = Dz + (long long)r * DIN_;
  float* xr = xs + (long long)r * DIN_;
  float s1 = 0.0f, s2 = 0.0f;
  for (int cIdx = tid; cIdx < DIN_; cIdx += 256) {
    float d = dz[cIdx], x = xr[cIdx];
    s1 += d * x; s2 += d * d;
  }
  r1[tid] = s1; r2[tid] = s2; __syncthreads();
  for (int s = 128; s > 0; s >>= 1) {
    if (tid < s) { r1[tid] += r1[tid + s]; r2[tid] += r2[tid + s]; }
    __syncthreads();
  }
  float nrm = sqrtf(r2[0]) + EPS_;
  float sc = r1[0] / (nrm * nrm);
  if (tid == 0) scale[r] = sc;
  for (int cIdx = tid; cIdx < DIN_; cIdx += 256) {
    float x = xr[cIdx] - sc * dz[cIdx];
    xr[cIdx] = x;
    xsbf[(long long)r * DIN_ + cIdx] = f2bf(x);
  }
}

__global__ void accum_zpred_kernel(const float* zpl, const float* scale, float* zpred) {
  int r = blockIdx.y;
  int cIdx = blockIdx.x * blockDim.x + threadIdx.x;
  long long off = (long long)r * W_ + cIdx;
  zpred[off] += zpl[off] * scale[r];
}

// Per-row top-32 of z_novel; zsum_bf = bf16(z_pred + masked z_novel)
__global__ __launch_bounds__(256) void topk_combine_kernel(
    const float* znov, const float* zpred, u16* zsum) {
  __shared__ float rowbuf[W_];               // 48 KB (<< 320 KB/WGP)
  __shared__ float rv[256];
  __shared__ int   ri[256];
  __shared__ float selv[TOPK_];
  __shared__ int   seli[TOPK_];
  int r = blockIdx.x, tid = threadIdx.x;
  const float* zn = znov + (long long)r * W_;
  const float* zp = zpred + (long long)r * W_;
  u16* out = zsum + (long long)r * W_;
  for (int cIdx = tid; cIdx < W_; cIdx += 256) rowbuf[cIdx] = zn[cIdx];
  __syncthreads();
  for (int t = 0; t < TOPK_; t++) {
    float bvv = -1e30f; int bi = 0;
    for (int cIdx = tid; cIdx < W_; cIdx += 256) {
      float v = rowbuf[cIdx];
      if (v > bvv) { bvv = v; bi = cIdx; }
    }
    rv[tid] = bvv; ri[tid] = bi; __syncthreads();
    for (int s = 128; s > 0; s >>= 1) {
      if (tid < s && rv[tid + s] > rv[tid]) { rv[tid] = rv[tid + s]; ri[tid] = ri[tid + s]; }
      __syncthreads();
    }
    if (tid == 0) { selv[t] = rv[0]; seli[t] = ri[0]; rowbuf[ri[0]] = -1e30f; }
    __syncthreads();
  }
  for (int cIdx = tid; cIdx < W_; cIdx += 256) out[cIdx] = f2bf(zp[cIdx]);
  __syncthreads();
  if (tid < TOPK_) {
    int cIdx = seli[tid];
    out[cIdx] = f2bf(zp[cIdx] + selv[tid]);
  }
}

// ---------------------------------------------------------------------------
// Host orchestration
// ---------------------------------------------------------------------------
static inline void launch_gemm(const u16* A, const u16* B, const float* bias,
                               float* Cf, u16* Cbf, int M, int N, int K,
                               long long sA, long long sB, long long sC, int batch,
                               float alpha, int relu, hipStream_t stream) {
  dim3 blk(256);
  if (N % 128 == 0) {
    dim3 g(N / 128, M / 128, batch);
    gemm_bf16_wmma_kernel<2><<<g, blk, 0, stream>>>(A, B, bias, Cf, Cbf, M, N, K,
                                                    sA, sB, sC, alpha, relu);
  } else {
    dim3 g(N / 64, M / 128, batch);
    gemm_bf16_wmma_kernel<1><<<g, blk, 0, stream>>>(A, B, bias, Cf, Cbf, M, N, K,
                                                    sA, sB, sC, alpha, relu);
  }
}

extern "C" void kernel_launch(void* const* d_in, const int* in_sizes, int n_in,
                              void* d_out, int out_size, void* d_ws, size_t ws_size,
                              hipStream_t stream) {
  (void)in_sizes; (void)n_in; (void)out_size; (void)ws_size;
  const float* x_in = (const float*)d_in[0];
  const float* D    = (const float*)d_in[1];
  const float* bvec = (const float*)d_in[2];
  const float* Wq   = (const float*)d_in[3];
  const float* bq   = (const float*)d_in[4];
  const float* Wk   = (const float*)d_in[5];
  const float* bk   = (const float*)d_in[6];
  const float* Wv   = (const float*)d_in[7];
  const float* bv   = (const float*)d_in[8];
  const float* Wo   = (const float*)d_in[9];
  const float* bo   = (const float*)d_in[10];
  float* out = (float*)d_out;

  size_t off = 0;
  auto alloc = [&](size_t bytes) -> void* {
    off = (off + 255) & ~(size_t)255;
    void* p = (char*)d_ws + off;
    off += bytes;
    return p;
  };
  u16*   DbfT  = (u16*)alloc((size_t)DIN_ * W_ * 2);   // E = D^T, [DIN,W]
  u16*   Dbf   = (u16*)alloc((size_t)W_ * DIN_ * 2);   // D, [W,DIN]
  u16*   Wqbf  = (u16*)alloc((size_t)W_ * DB_ * 2);
  u16*   Wkbf  = (u16*)alloc((size_t)W_ * DB_ * 2);
  u16*   Wvbf  = (u16*)alloc((size_t)W_ * DB_ * 2);
  u16*   Wobf  = (u16*)alloc((size_t)DB_ * W_ * 2);
  float* xs    = (float*)alloc((size_t)M_ * DIN_ * 4);
  u16*   xsbf  = (u16*)alloc((size_t)M_ * DIN_ * 2);
  u16*   zin   = (u16*)alloc((size_t)M_ * W_ * 2);
  u16*   zctx  = (u16*)alloc((size_t)M_ * W_ * 2);
  float* qf    = (float*)alloc((size_t)M_ * DB_ * 4);
  float* kf    = (float*)alloc((size_t)M_ * DB_ * 4);
  float* vf    = (float*)alloc((size_t)M_ * DB_ * 4);
  u16*   Qp    = (u16*)alloc((size_t)BH_ * L_ * HP_ * 2);
  u16*   Kt    = (u16*)alloc((size_t)BH_ * HP_ * L_ * 2);
  u16*   Vp    = (u16*)alloc((size_t)BH_ * L_ * VP_ * 2);
  float* scores= (float*)alloc((size_t)BH_ * L_ * L_ * 4);
  u16*   probs = (u16*)alloc((size_t)BH_ * L_ * L_ * 2);
  float* Opad  = (float*)alloc((size_t)BH_ * L_ * VP_ * 4);
  u16*   obf   = (u16*)alloc((size_t)M_ * DB_ * 2);
  float* zpl   = (float*)alloc((size_t)M_ * W_ * 4);   // z_pred_ (layer) f32
  u16*   zplbf = (u16*)alloc((size_t)M_ * W_ * 2);
  float* Dz    = (float*)alloc((size_t)M_ * DIN_ * 4);
  float* scl   = (float*)alloc((size_t)M_ * 4);
  float* zpred = (float*)alloc((size_t)M_ * W_ * 4);
  float* znov  = zpl;   // reuse after layer loop
  u16*   zsum  = zin;   // reuse after layer loop

  dim3 blk(256);

  // Prologue: weight conversion, residual init, z_pred = 0
  transpose_cvt_kernel<<<dim3(4096), blk, 0, stream>>>(D, DbfT, W_, DIN_);
  cvt_bf16_kernel<<<dim3(4096), blk, 0, stream>>>(D, Dbf, (long long)W_ * DIN_);
  fill_zero_kernel<<<dim3(4096), blk, 0, stream>>>(zpred, (long long)M_ * W_);
  prep_x_kernel<<<dim3(2048), blk, 0, stream>>>(x_in, bvec, xs, xsbf,
                                                (long long)M_ * DIN_, DIN_);

  for (int i = 0; i < NL_; i++) {
    // z_in = relu((x*LAM) @ E)
    launch_gemm(xsbf, DbfT, nullptr, nullptr, zin, M_, W_, DIN_,
                0, 0, 0, 1, LAM_, 1, stream);
    shift_rows_kernel<<<dim3(6, 4096), blk, 0, stream>>>(zin, zctx);

    // layer weights -> bf16
    cvt_bf16_kernel<<<dim3(2048), blk, 0, stream>>>(Wq + (size_t)i * W_ * DB_, Wqbf, (long long)W_ * DB_);
    cvt_bf16_kernel<<<dim3(2048), blk, 0, stream>>>(Wk + (size_t)i * W_ * DB_, Wkbf, (long long)W_ * DB_);
    cvt_bf16_kernel<<<dim3(2048), blk, 0, stream>>>(Wv + (size_t)i * W_ * DB_, Wvbf, (long long)W_ * DB_);
    cvt_bf16_kernel<<<dim3(2048), blk, 0, stream>>>(Wo + (size_t)i * DB_ * W_, Wobf, (long long)DB_ * W_);

    // q/k/v projections (N=192 -> NB=1 path)
    launch_gemm(zin,  Wqbf, bq + (size_t)i * DB_, qf, nullptr, M_, DB_, W_, 0, 0, 0, 1, 1.0f, 0, stream);
    launch_gemm(zctx, Wkbf, bk + (size_t)i * DB_, kf, nullptr, M_, DB_, W_, 0, 0, 0, 1, 1.0f, 0, stream);
    launch_gemm(zctx, Wvbf, bv + (size_t)i * DB_, vf, nullptr, M_, DB_, W_, 0, 0, 0, 1, 1.0f, 0, stream);
    pack_qkv_kernel<<<dim3((BH_ * L_ * VP_) / 256), blk, 0, stream>>>(qf, kf, vf, Qp, Kt, Vp);

    // scores = Q @ K^T (batched over 32 heads); softmax scale folded below
    launch_gemm(Qp, Kt, nullptr, scores, nullptr, L_, L_, HP_,
                (long long)L_ * HP_, (long long)HP_ * L_, (long long)L_ * L_,
                BH_, 1.0f, 0, stream);
    softmax_causal_kernel<<<dim3(L_, BH_), blk, 0, stream>>>(scores, probs);

    // O = probs @ V (batched, N=64 -> NB=1 path)
    launch_gemm(probs, Vp, nullptr, Opad, nullptr, L_, VP_, L_,
                (long long)L_ * L_, (long long)L_ * VP_, (long long)L_ * VP_,
                BH_, 1.0f, 0, stream);
    unpack_o_kernel<<<dim3((M_ * DB_ + 255) / 256), blk, 0, stream>>>(Opad, obf);

    // z_pred_ = relu(o @ Wo + bo)  (f32 + bf16 outputs)
    launch_gemm(obf, Wobf, bo + (size_t)i * W_, zpl, zplbf, M_, W_, DB_,
                0, 0, 0, 1, 1.0f, 1, stream);
    // Dz = z_pred_ @ D
    launch_gemm(zplbf, Dbf, nullptr, Dz, nullptr, M_, DIN_, W_,
                0, 0, 0, 1, 1.0f, 0, stream);
    // proj_scale, x update, z_pred accumulation
    proj_update_kernel<<<dim3(M_), blk, 0, stream>>>(Dz, xs, xsbf, scl);
    accum_zpred_kernel<<<dim3(W_ / 256, M_), blk, 0, stream>>>(zpl, scl, zpred);
  }

  // z_novel = relu((x*LAM) @ E) ; top-32 mask ; zsum = z_novel_masked + z_pred
  launch_gemm(xsbf, DbfT, nullptr, znov, nullptr, M_, W_, DIN_,
              0, 0, 0, 1, LAM_, 1, stream);
  topk_combine_kernel<<<dim3(M_), blk, 0, stream>>>(znov, zpred, zsum);

  // x_recons = zsum @ D + b
  launch_gemm(zsum, Dbf, bvec, out, nullptr, M_, DIN_, W_,
              0, 0, 0, 1, 1.0f, 0, stream);
}
